// SoftEquivariantLayer_28398323761421
// MI455X (gfx1250) — matrix operand order repaired and checked by reading, not compiled, not run on hardware
//
#include <hip/hip_runtime.h>
#include <hip/hip_bf16.h>
#include <math.h>

// ---------------------------------------------------------------------------
// SoftEquivariantLayer on MI455X (gfx1250, wave32, WMMA).
//
// Roofline: 1 KB HBM traffic per row, 43K flops per row -> ~42 flop/byte.
// At 23.3 TB/s the bandwidth floor is ~22us and requires ~1 PFLOP/s, which
// only the f16/bf16 WMMA path (16x16x32, f32 accum) can supply. All matmuls
// (the 128x128 mixing AND the tiny gating MLP) therefore run on
// v_wmma_f32_16x16x32_f16. gates=sigmoid(gate_bias) is folded into the
// mixing matrix columns at pack time.
// ---------------------------------------------------------------------------

typedef __attribute__((ext_vector_type(16))) _Float16 v16h;
typedef __attribute__((ext_vector_type(8)))  _Float16 v8h;
typedef __attribute__((ext_vector_type(8)))  float    v8f;

// Problem dims (fixed by the reference)
static constexpr int BATCH = 524288;
static constexpr int NB    = 8;
static constexpr int BD    = 16;
static constexpr int DIM   = NB * BD;   // 128
static constexpr int HID   = 64;
static constexpr int NT    = BATCH / 16; // 32768 row-tiles of 16

// Packed f16 weight layout in d_ws (halves), row strides padded so that
// per-lane row-strided B-fragment loads hit distinct LDS banks.
static constexpr int MH_OFF = 0;                 static constexpr int MH_LD = 136; // 128x128 (+gate folded)
static constexpr int W1_OFF = 128 * MH_LD;       static constexpr int W1_LD = 40;  // 64x32 (K padded 8->32)
static constexpr int W2_OFF = W1_OFF + 64*W1_LD; static constexpr int W2_LD = 72;  // 64x64
static constexpr int W3_OFF = W2_OFF + 64*W2_LD; static constexpr int W3_LD = 72;  // 16x64 (N padded 8->16)
static constexpr int WTOT_H = W3_OFF + 16*W3_LD; // 25728 halves = 51456 B

// Dynamic LDS layout (bytes)
static constexpr int SW_BYTES  = WTOT_H * 2;       // 51456
static constexpr int Z_LD      = 132;              // padded f32 row stride
static constexpr int H_LD      = 68;               // padded f32 row stride
static constexpr int ZB_OFF    = SW_BYTES;                   // 16*132*4 = 8448
static constexpr int HB_OFF    = ZB_OFF + 16*Z_LD*4;         // 16*68*4  = 4352
static constexpr int NB_OFF    = HB_OFF + 16*H_LD*4;         // 16*8*4   = 512
static constexpr int SB_OFF    = NB_OFF + 16*8*4;            // 16*8*4   = 512
static constexpr int SMEM_SIZE = SB_OFF + 16*8*4;            // 65280 B (< 64 KiB)

static __device__ __forceinline__ v8f fzero8() {
  v8f z;
  #pragma unroll
  for (int i = 0; i < 8; ++i) z[i] = 0.0f;
  return z;
}
static __device__ __forceinline__ v16h hzero16() {
  v16h z;
  #pragma unroll
  for (int i = 0; i < 16; ++i) z[i] = (_Float16)0.0f;
  return z;
}
static __device__ __forceinline__ float gelu_exact(float x) {
  return 0.5f * x * (1.0f + erff(x * 0.70710678118654752f));
}
static __device__ __forceinline__ float softplus_f(float x) {
  return fmaxf(x, 0.0f) + log1pf(expf(-fabsf(x)));
}

// A fragment (16x32, f16, M x K) from a row-major f32 LDS tile.
// ISA layout: lanes 0-15 hold row m=lane, halves 0..7 = K kbase+0..7,
// halves 8..15 = K kbase+16..23; lanes 16-31 the +8 K offsets.
static __device__ __forceinline__ v16h ldA_f32(const float* buf, int ldw,
                                               int kbase, int lane) {
  const int m  = lane & 15;
  const int kh = lane >> 4;
  const float4* g0 = (const float4*)&buf[m * ldw + kbase + 8 * kh];
  const float4* g1 = (const float4*)&buf[m * ldw + kbase + 16 + 8 * kh];
  float4 x0 = g0[0], x1 = g0[1], y0 = g1[0], y1 = g1[1];
  v16h a;
  a[0]=(_Float16)x0.x; a[1]=(_Float16)x0.y; a[2]=(_Float16)x0.z; a[3]=(_Float16)x0.w;
  a[4]=(_Float16)x1.x; a[5]=(_Float16)x1.y; a[6]=(_Float16)x1.z; a[7]=(_Float16)x1.w;
  a[8]=(_Float16)y0.x; a[9]=(_Float16)y0.y; a[10]=(_Float16)y0.z; a[11]=(_Float16)y0.w;
  a[12]=(_Float16)y1.x; a[13]=(_Float16)y1.y; a[14]=(_Float16)y1.z; a[15]=(_Float16)y1.w;
  return a;
}

// B fragment (32x16, f16, K x N) from LDS stored as [N][K] f16 (row = output
// column). ISA layout: lane n=lane&15 holds column n, halves 0..7 = K
// kbase+8*(lane>>4)+0..7, halves 8..15 = K kbase+16+8*(lane>>4)+0..7.
// Two 16B loads (rows only guaranteed 16B aligned).
static __device__ __forceinline__ v16h ldB(const _Float16* W, int ldk,
                                           int nbase, int kbase, int lane) {
  const int n = nbase + (lane & 15);
  const int k = kbase + 8 * (lane >> 4);
  v8h lo = *(const v8h*)&W[n * ldk + k];
  v8h hi = *(const v8h*)&W[n * ldk + k + 16];
  return __builtin_shufflevector(lo, hi, 0,1,2,3,4,5,6,7,8,9,10,11,12,13,14,15);
}

// ---------------------------------------------------------------------------
// Pack kernel: build f16 weight block in d_ws.
//   Mh[n=16i+k][kk=16j+d] = mix_w[i,j,k,d] * sigmoid(gate_bias[i])
//   W1h[n][k] = w1[n][k] (k<8 else 0);  W2h = w2;  W3h[n][k] = w3 (n<8 else 0)
// ---------------------------------------------------------------------------
__global__ void pack_weights_kernel(const float* __restrict__ w1,
                                    const float* __restrict__ w2,
                                    const float* __restrict__ w3,
                                    const float* __restrict__ mix_w,
                                    const float* __restrict__ gate_bias,
                                    _Float16* __restrict__ wp) {
  int idx = blockIdx.x * blockDim.x + threadIdx.x;
  if (idx < 16384) {                       // Mh: 128x128
    int n = idx >> 7, kk = idx & 127;
    int i = n >> 4, k = n & 15, j = kk >> 4, d = kk & 15;
    float g = 1.0f / (1.0f + expf(-gate_bias[i]));
    float v = mix_w[(((i * 8) + j) * 16 + k) * 16 + d] * g;
    wp[MH_OFF + n * MH_LD + kk] = (_Float16)v;
  } else if (idx < 16384 + 2048) {         // W1: 64x32 (K-pad)
    int r = idx - 16384;
    int n = r >> 5, k = r & 31;
    float v = (k < 8) ? w1[n * 8 + k] : 0.0f;
    wp[W1_OFF + n * W1_LD + k] = (_Float16)v;
  } else if (idx < 16384 + 2048 + 4096) {  // W2: 64x64
    int r = idx - 18432;
    int n = r >> 6, k = r & 63;
    wp[W2_OFF + n * W2_LD + k] = (_Float16)w2[n * 64 + k];
  } else if (idx < 16384 + 2048 + 4096 + 1024) { // W3: 16x64 (N-pad)
    int r = idx - 22528;
    int n = r >> 6, k = r & 63;
    float v = (n < 8) ? w3[n * 64 + k] : 0.0f;
    wp[W3_OFF + n * W3_LD + k] = (_Float16)v;
  }
}

// ---------------------------------------------------------------------------
// Main kernel: one wave32 per workgroup; grid-stride over 16-row tiles.
// ---------------------------------------------------------------------------
__global__ __launch_bounds__(32)
void soft_equivariant_kernel(const float* __restrict__ z,
                             const float* __restrict__ b1,
                             const float* __restrict__ b2,
                             const _Float16* __restrict__ wp,
                             float* __restrict__ out) {
  extern __shared__ char smem[];
  _Float16* sW   = (_Float16*)smem;
  float*    zbuf = (float*)(smem + ZB_OFF);
  float*    hbuf = (float*)(smem + HB_OFF);
  float*    nbuf = (float*)(smem + NB_OFF);
  float*    sbuf = (float*)(smem + SB_OFF);

  const int lane = threadIdx.x & 31;
  const int nlo  = lane & 15;
  const int kh   = lane >> 4;

  // Stage all packed weights into LDS once (shared across all tiles).
  {
    const uint4* src = (const uint4*)wp;
    uint4*       dst = (uint4*)sW;
    for (int i = lane; i < (SW_BYTES / 16); i += 32) dst[i] = src[i];
  }
  __syncthreads();

  const _Float16* sMh = sW + MH_OFF;
  const _Float16* sW1 = sW + W1_OFF;
  const _Float16* sW2 = sW + W2_OFF;
  const _Float16* sW3 = sW + W3_OFF;

  const int stride = (int)gridDim.x;
  for (int t = blockIdx.x; t < NT; t += stride) {
    const long row0 = (long)t * 16;
    const float* zt = z + row0 * DIM;

    // Prefetch next tile of z into cache (global_prefetch_b8).
    if (t + stride < NT) {
      const float* nz = z + (row0 + (long)stride * 16) * DIM;
      __builtin_prefetch(nz + lane * 64, 0, 1);
      __builtin_prefetch(nz + lane * 64 + 32, 0, 1);
    }

    // ---- stage z tile: 16 x 128 f32 -> LDS (padded stride) -------------
    for (int i = lane; i < 512; i += 32) {
      int r = i >> 5, c4 = i & 31;
      float4 v = ((const float4*)zt)[(r << 5) + c4];
      *(float4*)&zbuf[r * Z_LD + c4 * 4] = v;
    }
    __syncthreads();

    // ---- bundle norms: nbuf[r][j] = ||z[r, 16j..16j+15]|| + 1e-8 -------
    {
      const int r = nlo;
      #pragma unroll
      for (int jj = 0; jj < 4; ++jj) {
        int j = jj + 4 * kh;
        const float* zr = &zbuf[r * Z_LD + j * 16];
        float s = 0.0f;
        #pragma unroll
        for (int d = 0; d < 16; ++d) { float v = zr[d]; s = fmaf(v, v, s); }
        nbuf[r * 8 + j] = sqrtf(s) + 1e-8f;
      }
    }
    __syncthreads();

    // ---- MLP layer 1: h1 = gelu(norms @ W1^T + b1), K padded 8->32 -----
    v16h a1 = hzero16();
    if (kh == 0) {
      #pragma unroll
      for (int i = 0; i < 8; ++i) a1[i] = (_Float16)nbuf[nlo * 8 + i];
    }
    #pragma unroll
    for (int nt = 0; nt < 4; ++nt) {
      v8f d = fzero8();
      d = __builtin_amdgcn_wmma_f32_16x16x32_f16(
          false, a1, false, ldB(sW1, W1_LD, nt * 16, 0, lane),
          (short)0, d, false, false);
      float bias = b1[nt * 16 + nlo];
      #pragma unroll
      for (int v = 0; v < 8; ++v)
        hbuf[(v + 8 * kh) * H_LD + nt * 16 + nlo] = gelu_exact(d[v] + bias);
    }
    __syncthreads();

    // ---- MLP layer 2: h2 = gelu(h1 @ W2^T + b2) ------------------------
    v16h a20 = ldA_f32(hbuf, H_LD, 0, lane);
    v16h a21 = ldA_f32(hbuf, H_LD, 32, lane);
    __syncthreads();  // all lanes read h1 before h2 overwrites hbuf
    #pragma unroll
    for (int nt = 0; nt < 4; ++nt) {
      v8f d = fzero8();
      d = __builtin_amdgcn_wmma_f32_16x16x32_f16(
          false, a20, false, ldB(sW2, W2_LD, nt * 16, 0, lane),
          (short)0, d, false, false);
      d = __builtin_amdgcn_wmma_f32_16x16x32_f16(
          false, a21, false, ldB(sW2, W2_LD, nt * 16, 32, lane),
          (short)0, d, false, false);
      float bias = b2[nt * 16 + nlo];
      #pragma unroll
      for (int v = 0; v < 8; ++v)
        hbuf[(v + 8 * kh) * H_LD + nt * 16 + nlo] = gelu_exact(d[v] + bias);
    }
    __syncthreads();

    // ---- MLP layer 3: scales = softplus(h2 @ W3^T), N padded 8->16 -----
    {
      v16h a30 = ldA_f32(hbuf, H_LD, 0, lane);
      v16h a31 = ldA_f32(hbuf, H_LD, 32, lane);
      v8f d = fzero8();
      d = __builtin_amdgcn_wmma_f32_16x16x32_f16(
          false, a30, false, ldB(sW3, W3_LD, 0, 0, lane),
          (short)0, d, false, false);
      d = __builtin_amdgcn_wmma_f32_16x16x32_f16(
          false, a31, false, ldB(sW3, W3_LD, 0, 32, lane),
          (short)0, d, false, false);
      if (nlo < 8) {
        #pragma unroll
        for (int v = 0; v < 8; ++v)
          sbuf[(v + 8 * kh) * 8 + nlo] = softplus_f(d[v]);
      }
    }
    __syncthreads();

    // ---- mixing: [16x128] @ [128x128] (gates folded in), fused epilogue -
    v16h am0 = ldA_f32(zbuf, Z_LD, 0, lane);
    v16h am1 = ldA_f32(zbuf, Z_LD, 32, lane);
    v16h am2 = ldA_f32(zbuf, Z_LD, 64, lane);
    v16h am3 = ldA_f32(zbuf, Z_LD, 96, lane);
    #pragma unroll
    for (int nt = 0; nt < 8; ++nt) {
      v8f d = fzero8();
      d = __builtin_amdgcn_wmma_f32_16x16x32_f16(
          false, am0, false, ldB(sMh, MH_LD, nt * 16, 0, lane),
          (short)0, d, false, false);
      d = __builtin_amdgcn_wmma_f32_16x16x32_f16(
          false, am1, false, ldB(sMh, MH_LD, nt * 16, 32, lane),
          (short)0, d, false, false);
      d = __builtin_amdgcn_wmma_f32_16x16x32_f16(
          false, am2, false, ldB(sMh, MH_LD, nt * 16, 64, lane),
          (short)0, d, false, false);
      d = __builtin_amdgcn_wmma_f32_16x16x32_f16(
          false, am3, false, ldB(sMh, MH_LD, nt * 16, 96, lane),
          (short)0, d, false, false);
      const int col = nt * 16 + nlo;
      float* obase = out + row0 * DIM + col;
      #pragma unroll
      for (int v = 0; v < 8; ++v) {
        int r = v + 8 * kh;
        float zv = zbuf[r * Z_LD + col];
        float sc = sbuf[r * 8 + nt];       // bundle index == nt for this col
        obase[r * DIM] = fmaf(zv, sc, zv + d[v]);  // z + z*s + gated mixing
      }
    }
    __syncthreads();  // protect zbuf/nbuf before next tile's staging
  }
}

extern "C" void kernel_launch(void* const* d_in, const int* in_sizes, int n_in,
                              void* d_out, int out_size, void* d_ws, size_t ws_size,
                              hipStream_t stream) {
  (void)in_sizes; (void)n_in; (void)out_size; (void)ws_size;
  const float* z    = (const float*)d_in[0];
  const float* w1   = (const float*)d_in[1];
  const float* b1   = (const float*)d_in[2];
  const float* w2   = (const float*)d_in[3];
  const float* b2   = (const float*)d_in[4];
  const float* w3   = (const float*)d_in[5];
  const float* mixw = (const float*)d_in[6];
  const float* gb   = (const float*)d_in[7];
  float*       out  = (float*)d_out;
  _Float16*    wp   = (_Float16*)d_ws;

  // 23552 logical weight elements -> packed/padded f16 block in d_ws.
  pack_weights_kernel<<<92, 256, 0, stream>>>(w1, w2, w3, mixw, gb, wp);

  // 2048 single-wave workgroups, 16 tiles each (32768 tiles total).
  soft_equivariant_kernel<<<2048, 32, SMEM_SIZE, stream>>>(z, b1, b2, wp, out);
}
